// MarginalNetwork_9431748182755
// MI455X (gfx1250) — matrix operand order repaired and checked by reading, not compiled, not run on hardware
//
#include <hip/hip_runtime.h>
#include <math.h>

// ---------------------------------------------------------------------------
// MarginalNetwork forward for MI455X (gfx1250, wave32, WMMA).
//
// Input flattening assumption (setup_inputs dict order, nested params
// flattened in sorted-key pytree order):
//   0:x(8,512,10) 1:adj(8,512,512) 2:t(8)
//   3..8:   final_adj  W1(22,44) W2(44,44) W3(44,1) b1 b2 b3
//   9..14:  final_x    W1(32,64) W2(64,64) W3(64,10) b1 b2 b3
//   15+14l: layer l:   W1 W2 Wk Wm1 Wm2 Wq Wv b1 b2 bk bm1 bm2 bq bv
// Layers: (C,in,c_out) = (2,20,8), (8,32,8), (8,32,4); hid=16 everywhere.
// Output: x_out (8,512,10) then adj_pred (8,512,512), flat f32.
// Workspace requirement: ~346 MB (bump-allocated below).
//
// WMMA fragment layouts per ISA 7.12.2: a lane's 16-bit A fragment is two
// contiguous 8-element chunks (k in [g*8,g*8+8) u [g*8+16,g*8+24)), so LDS
// tiles are stored so fragments load as two ds_load_b128, with an 8-chunk
// XOR swizzle on (row>>2)&3 to break the stride-64B bank-conflict pattern.
// ---------------------------------------------------------------------------

typedef __attribute__((ext_vector_type(16))) __bf16 v16bf;
typedef __attribute__((ext_vector_type(8)))  __bf16 v8bf;
typedef __attribute__((ext_vector_type(4)))  __bf16 v4bf;
typedef __attribute__((ext_vector_type(8)))  float  v8f;
typedef __attribute__((ext_vector_type(4)))  float  v4f;
typedef __attribute__((ext_vector_type(2)))  float  v2f;

#if defined(__has_builtin)
#  if __has_builtin(__builtin_amdgcn_wmma_f32_16x16x4_f32)
#    define HAVE_WMMA_F32X4 1
#  endif
#endif
#ifndef HAVE_WMMA_F32X4
#  define HAVE_WMMA_F32X4 0
#endif

static __device__ __forceinline__ v8f wmma_bf16(v16bf a, v16bf b, v8f c) {
    return __builtin_amdgcn_wmma_f32_16x16x32_bf16(false, a, false, b, (short)0, c, false, false);
}
static __device__ __forceinline__ v16bf cat16(v8bf lo, v8bf hi) {
    return __builtin_shufflevector(lo, hi, 0,1,2,3,4,5,6,7,8,9,10,11,12,13,14,15);
}
static __device__ __forceinline__ float eluf(float z) { return z > 0.f ? z : (expf(z) - 1.f); }

// ------------------------- small elementwise kernels -------------------------

__global__ void k_build_x0(const float* __restrict__ x, const float* __restrict__ t,
                           float* __restrict__ x0)
{
    int idx = blockIdx.x * blockDim.x + threadIdx.x;     // (b*512+n)
    if (idx >= 8 * 512) return;
    int b = idx >> 9;
    float tv = t[b];
    const float* xp = x + (long)idx * 10;
    float* o = x0 + (long)idx * 20;
    for (int f = 0; f < 10; ++f) o[f] = xp[f];
    for (int i = 0; i < 5; ++i) {
        float fr = expf(-logf(10000.f) * (float)i / 5.f);
        o[10 + i] = cosf(tv * fr);
        o[15 + i] = sinf(tv * fr);
    }
}

__global__ void k_copy_ch0(const float* __restrict__ adj, float* __restrict__ adjc0)
{
    long idx = (long)blockIdx.x * blockDim.x + threadIdx.x;   // B*N*N
    if (idx >= 8L * 512 * 512) return;
    long b = idx >> 18, rem = idx & 262143;
    adjc0[b * 2 * 262144 + rem] = adj[idx];
}

__global__ void k_degree(const float* __restrict__ adjc, float* __restrict__ deg, int BC)
{
    int idx = blockIdx.x * blockDim.x + threadIdx.x;          // (bc*512+n)
    if (idx >= BC * 512) return;
    int n = idx & 511;
    const float* row = adjc + ((long)(idx >> 9) << 18) + ((long)n << 9);
    float s = 0.f;
    for (int m = 0; m < 512; ++m) s += (m == n) ? 1.f : row[m];
    deg[idx] = rsqrtf(fmaxf(s, 1.f));                         // clip(sum,1)^-0.5
}

__global__ void k_normadj(const float* __restrict__ adjc, const float* __restrict__ deg,
                          float* __restrict__ outn, long total)
{
    long idx = (long)blockIdx.x * blockDim.x + threadIdx.x;
    if (idx >= total) return;
    int m = (int)(idx & 511), n = (int)((idx >> 9) & 511);
    long bc = idx >> 18;
    float a = (m == n) ? 1.f : adjc[idx];
    outn[idx] = deg[(bc << 9) + n] * a * deg[(bc << 9) + m];
}

__global__ void k_sym(const float* __restrict__ in, float* __restrict__ out,
                      float scale, long total)
{
    long idx = (long)blockIdx.x * blockDim.x + threadIdx.x;
    if (idx >= total) return;
    int m = (int)(idx & 511), n = (int)((idx >> 9) & 511);
    long bc = idx >> 18;
    out[idx] = scale * (in[idx] + in[(bc << 18) + ((long)m << 9) + n]);
}

// h[b,c,n,d] = sum_f x[b,n,f] * W[c,f,d] + bias[c,d]
__global__ void k_linear(const float* __restrict__ x, const float* __restrict__ W,
                         const float* __restrict__ bias, float* __restrict__ h,
                         int C, int in, int total)
{
    int idx = blockIdx.x * blockDim.x + threadIdx.x;
    if (idx >= total) return;
    int d = idx & 31, n = (idx >> 5) & 511, bc = idx >> 14;
    int b = bc / C, c = bc - b * C;
    const float* xp = x + ((long)b * 512 + n) * in;
    const float* wp = W + (long)c * in * 32 + d;
    float s = bias[c * 32 + d];
    for (int f = 0; f < in; ++f) s += xp[f] * wp[f * 32];
    h[idx] = s;
}

// --------------------- LDS-staged bf16 WMMA batched GEMM ---------------------
// Grid: (M/64, N/16, batch), block 128 (4 waves), one 16x16 tile per wave.
// A tile (64x32): row-major bf16, 8-elt chunks XOR-swizzled by (row>>2)&3.
// B tile (32x16): n-major bf16, same chunk swizzle on (n>>2)&3.
// Fragments: two ds_load_b128 per operand.  lda must be a multiple of 4.

__global__ __launch_bounds__(128)
void k_gemm_bf16(const float* __restrict__ A, const float* __restrict__ B,
                 float* __restrict__ C, int K, int lda, int ldb, int ldc,
                 long sA, long sB, long sC)
{
    __shared__ __align__(16) __bf16 As[64 * 32];
    __shared__ __align__(16) __bf16 Bs[16 * 32];
    const int tid  = threadIdx.x;
    const int wave = tid >> 5, lane = tid & 31;
    const int g = lane >> 4, ln = lane & 15;
    const int m0 = blockIdx.x * 64, n0 = blockIdx.y * 16;
    const float* Ap = A + (long)blockIdx.z * sA;
    const float* Bp = B + (long)blockIdx.z * sB;
    float*       Cp = C + (long)blockIdx.z * sC;

    v8f acc = {};
    for (int k0 = 0; k0 < K; k0 += 32) {
        // stage A: 512 float4 groups, 4 per thread; packed bf16x4 ds_store_b64
        for (int i = 0; i < 4; ++i) {
            int gi = i * 128 + tid;
            int r = gi >> 3, kq = (gi & 7) * 4;
            v4f av = *(const v4f*)(Ap + (long)(m0 + r) * lda + k0 + kq);
            v4bf ab;
            ab[0] = (__bf16)av[0]; ab[1] = (__bf16)av[1];
            ab[2] = (__bf16)av[2]; ab[3] = (__bf16)av[3];
            int sw = (r >> 2) & 3;
            *(v4bf*)(As + r * 32 + (((kq >> 3) ^ sw) << 3) + (kq & 7)) = ab;
        }
        if (k0 + 32 < K)
            __builtin_prefetch(Ap + (long)(m0 + (tid >> 1)) * lda + k0 + 32 + (tid & 1) * 16, 0, 0);
        // stage B: 32x16, one float4 along n per thread, scattered to n-major
        {
            int k = tid >> 2, nq = (tid & 3) * 4;
            v4f bv = *(const v4f*)(Bp + (long)(k0 + k) * ldb + n0 + nq);
            for (int u = 0; u < 4; ++u) {
                int n = nq + u, sw = (n >> 2) & 3;
                Bs[n * 32 + (((k >> 3) ^ sw) << 3) + (k & 7)] = (__bf16)bv[u];
            }
        }
        __syncthreads();
        const int rr  = wave * 16 + ln;
        const int swa = (rr >> 2) & 3;
        const __bf16* ar = As + rr * 32;
        v8bf a0 = *(const v8bf*)(ar + ((g ^ swa) << 3));
        v8bf a1 = *(const v8bf*)(ar + (((g + 2) ^ swa) << 3));
        const int swb = (ln >> 2) & 3;
        const __bf16* br = Bs + ln * 32;
        v8bf b0 = *(const v8bf*)(br + ((g ^ swb) << 3));
        v8bf b1 = *(const v8bf*)(br + (((g + 2) ^ swb) << 3));
        acc = wmma_bf16(cat16(a0, a1), cat16(b0, b1), acc);
        __syncthreads();
    }
    for (int v = 0; v < 8; ++v)   // C: M = v + 8g, N = lane&15
        Cp[(long)(m0 + wave * 16 + v + 8 * g) * ldc + n0 + ln] = acc[v];
}

// ------------- attention scores: tanh(QK^T/sqrt32) head-mean -----------------
// One wave per 16x16 tile; per head two V_WMMA_F32_16X16X4_F32 (K=8, f32).

__global__ __launch_bounds__(32)
void k_scores(const float* __restrict__ Q, const float* __restrict__ Km,
              float* __restrict__ S)
{
    const int bc = blockIdx.z;
    const int n0 = blockIdx.x * 16, m0 = blockIdx.y * 16;
    const int lane = threadIdx.x & 31, g = lane >> 4, ln = lane & 15;
    const float inv = 0.1767766953f;                 // 1/sqrt(32)
    const float* Qp = Q  + (long)bc * 512 * 32;
    const float* Kp = Km + (long)bc * 512 * 32;
    float accv[8] = {0, 0, 0, 0, 0, 0, 0, 0};
#if HAVE_WMMA_F32X4
    for (int h = 0; h < 4; ++h) {
        v8f c = {};
        for (int tctr = 0; tctr < 2; ++tctr) {
            int kb = h * 8 + tctr * 4 + g * 2;       // f32 A 16x4: lane group picks K pair
            v2f a  = *(const v2f*)(Qp + (n0 + ln) * 32 + kb);
            v2f bv = *(const v2f*)(Kp + (m0 + ln) * 32 + kb);
            c = __builtin_amdgcn_wmma_f32_16x16x4_f32(false, a, false, bv,
                                                      (short)0, c, false, false);
        }
        for (int v = 0; v < 8; ++v) accv[v] += tanhf(c[v] * inv);
    }
#else
    for (int v = 0; v < 8; ++v) {
        int n = n0 + v + 8 * g, m = m0 + ln;
        for (int h = 0; h < 4; ++h) {
            float s = 0.f;
            for (int d = 0; d < 8; ++d) s += Qp[n * 32 + h * 8 + d] * Kp[m * 32 + h * 8 + d];
            accv[v] += tanhf(s * inv);
        }
    }
#endif
    for (int v = 0; v < 8; ++v)
        S[((long)bc * 512 + n0 + v + 8 * g) * 512 + m0 + ln] = accv[v] * 0.25f;
}

// ---------------------- per-node / per-edge small MLPs -----------------------

__global__ void k_xmlp(const float* __restrict__ V, const float* __restrict__ Wm1,
                       const float* __restrict__ bm1, const float* __restrict__ Wm2,
                       const float* __restrict__ bm2, float* __restrict__ xo, int C)
{
    int idx = blockIdx.x * blockDim.x + threadIdx.x;      // (b*512+n)
    if (idx >= 8 * 512) return;
    int b = idx >> 9, n = idx & 511;
    float h[16];
    for (int j = 0; j < 16; ++j) h[j] = bm1[j];
    for (int c = 0; c < C; ++c) {
        const float* vp = V + (((long)b * C + c) * 512 + n) * 32;
        for (int d = 0; d < 32; ++d) {
            float vv = vp[d];
            const float* wr = Wm1 + (c * 32 + d) * 16;
            for (int j = 0; j < 16; ++j) h[j] += vv * wr[j];
        }
    }
    for (int j = 0; j < 16; ++j) h[j] = eluf(h[j]);
    float* op = xo + (long)idx * 32;
    for (int d = 0; d < 32; ++d) {
        float s = bm2[d];
        for (int j = 0; j < 16; ++j) s += h[j] * Wm2[j * 32 + d];
        op[d] = tanhf(s);
    }
}

__global__ void k_adjmlp(const float* __restrict__ A, const float* __restrict__ adjc,
                         const float* __restrict__ W1, const float* __restrict__ b1,
                         const float* __restrict__ W2, const float* __restrict__ b2,
                         float* __restrict__ tmp, int C, int cout)
{
    long idx = (long)blockIdx.x * blockDim.x + threadIdx.x;   // B*N*N
    if (idx >= 8L * 512 * 512) return;
    int b = (int)(idx >> 18), rm = (int)(idx & 262143);
    int n = rm >> 9, m = rm & 511;
    float f[16];
    for (int c = 0; c < C; ++c) {
        long e = (((long)b * C + c) * 512 + n) * 512 + m;
        f[c]     = A[e];
        f[C + c] = adjc[e];
    }
    float o[8];
    for (int co = 0; co < cout; ++co) o[co] = b2[co];
    for (int j = 0; j < 16; ++j) {
        float hv = b1[j];
        for (int i = 0; i < 2 * C; ++i) hv += f[i] * W1[i * 16 + j];
        hv = eluf(hv);
        for (int co = 0; co < cout; ++co) o[co] += hv * W2[j * cout + co];
    }
    for (int co = 0; co < cout; ++co)
        tmp[(((long)b * cout + co) * 512 + n) * 512 + m] = o[co];
}

// ----------- fused final adjacency head: 22->44->44->1, WMMA, 12.5 GF --------
// Each wave pushes 16 rows through 22->44->44->1. K padded 22->32 / 44->64,
// N padded 44->48 (3 tiles). Weights staged n-major in LDS so B fragments are
// contiguous (2x ds_load_b128); inter-layer activations kept in per-wave
// bf16 LDS rows padded to K=64 so layer-2 A fragments load the same way.

__global__ __launch_bounds__(128)
void k_final_adj(const float* __restrict__ ch0, const float* __restrict__ ch1,
                 const float* __restrict__ ch2, const float* __restrict__ ch3,
                 const float* __restrict__ W1, const float* __restrict__ b1,
                 const float* __restrict__ W2, const float* __restrict__ b2,
                 const float* __restrict__ W3, const float* __restrict__ b3,
                 float* __restrict__ out)
{
    __shared__ __align__(16) __bf16 W1f[48 * 32];   // [n][k], K padded to 32
    __shared__ __align__(16) __bf16 W2f[48 * 64];   // [n][k], K padded to 64
    __shared__ __align__(16) __bf16 Hb[4][16][64];  // per-wave activations
    __shared__ float W3L[48], b1L[48], b2L[48];

    const int tid = threadIdx.x;
    for (int i = tid; i < 48 * 32; i += 128) {
        int n = i >> 5, k = i & 31;
        W1f[i] = (__bf16)((n < 44 && k < 22) ? W1[k * 44 + n] : 0.f);
    }
    for (int i = tid; i < 48 * 64; i += 128) {
        int n = i >> 6, k = i & 63;
        W2f[i] = (__bf16)((n < 44 && k < 44) ? W2[k * 44 + n] : 0.f);
    }
    for (int i = tid; i < 48; i += 128) {
        b1L[i] = (i < 44) ? b1[i] : 0.f;
        b2L[i] = (i < 44) ? b2[i] : 0.f;
        W3L[i] = (i < 44) ? W3[i] : 0.f;
    }
    __syncthreads();

    const int wave = tid >> 5, lane = tid & 31;
    const int g = lane >> 4, ln = lane & 15;
    const long r = ((long)blockIdx.x * 4 + wave) * 16 + ln;   // this lane's row
    const int b  = (int)(r >> 18);
    const int rm = (int)(r & 262143);
    const int n  = rm >> 9, m = rm & 511;

    // zero the K-pad region of this wave's activation rows (cols 48..63)
    for (int v = 0; v < 8; ++v) Hb[wave][v + 8 * g][48 + ln] = (__bf16)0.f;

    // layer 1: gather 22-feature row from the stacked adjacency channels
    v16bf af;
    for (int v = 0; v < 8; ++v)
        for (int hh = 0; hh < 2; ++hh) {
            int j = 2 * v + hh;
            int k = g * 8 + ((v >= 4) ? 16 : 0) + 2 * (v & 3) + hh;
            float val = 0.f;
            if (k < 2)       val = ch0[(((long)b * 2 + k     ) * 512 + n) * 512 + m];
            else if (k < 10) val = ch1[(((long)b * 8 + (k-2) ) * 512 + n) * 512 + m];
            else if (k < 18) val = ch2[(((long)b * 8 + (k-10)) * 512 + n) * 512 + m];
            else if (k < 22) val = ch3[(((long)b * 4 + (k-18)) * 512 + n) * 512 + m];
            af[j] = (__bf16)val;
        }
    v8f acc[3] = {};
    for (int nt = 0; nt < 3; ++nt) {
        const __bf16* wr = W1f + (nt * 16 + ln) * 32 + g * 8;
        acc[nt] = wmma_bf16(af, cat16(*(const v8bf*)wr, *(const v8bf*)(wr + 16)), acc[nt]);
    }
    for (int nt = 0; nt < 3; ++nt)
        for (int v = 0; v < 8; ++v) {
            int col = nt * 16 + ln;
            Hb[wave][v + 8 * g][col] = (__bf16)eluf(acc[nt][v] + b1L[col]);
        }
    asm volatile("s_wait_dscnt 0x0" ::: "memory");   // wave-private LDS RAW

    // layer 2: 44 -> 44 (2 K-steps of 32); A fragments straight from Hb
    v16bf a2[2];
    {
        const __bf16* hr = &Hb[wave][ln][0];
        for (int ks = 0; ks < 2; ++ks)
            a2[ks] = cat16(*(const v8bf*)(hr + ks * 32 + g * 8),
                           *(const v8bf*)(hr + ks * 32 + g * 8 + 16));
    }
    v8f d[3] = {};
    for (int nt = 0; nt < 3; ++nt)
        for (int ks = 0; ks < 2; ++ks) {
            const __bf16* wr = W2f + (nt * 16 + ln) * 64 + ks * 32 + g * 8;
            d[nt] = wmma_bf16(a2[ks], cat16(*(const v8bf*)wr, *(const v8bf*)(wr + 16)), d[nt]);
        }
    for (int nt = 0; nt < 3; ++nt)
        for (int v = 0; v < 8; ++v) {
            int col = nt * 16 + ln;
            Hb[wave][v + 8 * g][col] = (__bf16)eluf(d[nt][v] + b2L[col]);
        }
    asm volatile("s_wait_dscnt 0x0" ::: "memory");

    // layer 3: N=1 -> plain dot, sigmoid, zero diagonal
    if (lane < 16) {
        float s = b3[0];
        for (int j = 0; j < 44; ++j) s += W3L[j] * (float)Hb[wave][lane][j];
        float p = 1.f / (1.f + expf(-s));
        out[r] = (n == m) ? 0.f : p;
    }
}

__global__ void k_final_x(const float* __restrict__ x, const float* __restrict__ W1,
                          const float* __restrict__ b1, const float* __restrict__ W2,
                          const float* __restrict__ b2, const float* __restrict__ W3,
                          const float* __restrict__ b3, float* __restrict__ out)
{
    int idx = blockIdx.x * blockDim.x + threadIdx.x;   // (b*512+n)
    if (idx >= 8 * 512) return;
    const float* xp = x + (long)idx * 32;
    float h1[64], h2[64];
    for (int j = 0; j < 64; ++j) {
        float s = b1[j];
        for (int f = 0; f < 32; ++f) s += xp[f] * W1[f * 64 + j];
        h1[j] = eluf(s);
    }
    for (int j = 0; j < 64; ++j) {
        float s = b2[j];
        for (int i = 0; i < 64; ++i) s += h1[i] * W2[i * 64 + j];
        h2[j] = eluf(s);
    }
    for (int f = 0; f < 10; ++f) {
        float s = b3[f];
        for (int i = 0; i < 64; ++i) s += h2[i] * W3[i * 10 + f];
        out[(long)idx * 10 + f] = s;
    }
}

// --------------------------------- driver ------------------------------------

extern "C" void kernel_launch(void* const* d_in, const int* in_sizes, int n_in,
                              void* d_out, int out_size, void* d_ws, size_t ws_size,
                              hipStream_t stream)
{
    (void)in_sizes; (void)n_in; (void)out_size; (void)ws_size;
    const float* x    = (const float*)d_in[0];
    const float* adj  = (const float*)d_in[1];
    const float* tt   = (const float*)d_in[2];
    const float* faW1 = (const float*)d_in[3];
    const float* faW2 = (const float*)d_in[4];
    const float* faW3 = (const float*)d_in[5];
    const float* fab1 = (const float*)d_in[6];
    const float* fab2 = (const float*)d_in[7];
    const float* fab3 = (const float*)d_in[8];
    const float* fxW1 = (const float*)d_in[9];
    const float* fxW2 = (const float*)d_in[10];
    const float* fxW3 = (const float*)d_in[11];
    const float* fxb1 = (const float*)d_in[12];
    const float* fxb2 = (const float*)d_in[13];
    const float* fxb3 = (const float*)d_in[14];

    struct LP { const float *W1,*W2,*Wk,*Wm1,*Wm2,*Wq,*Wv,*b1,*b2,*bk,*bm1,*bm2,*bq,*bv; };
    LP L[3];
    for (int l = 0; l < 3; ++l) {
        int s = 15 + 14 * l;
        L[l].W1  = (const float*)d_in[s + 0];  L[l].W2  = (const float*)d_in[s + 1];
        L[l].Wk  = (const float*)d_in[s + 2];  L[l].Wm1 = (const float*)d_in[s + 3];
        L[l].Wm2 = (const float*)d_in[s + 4];  L[l].Wq  = (const float*)d_in[s + 5];
        L[l].Wv  = (const float*)d_in[s + 6];  L[l].b1  = (const float*)d_in[s + 7];
        L[l].b2  = (const float*)d_in[s + 8];  L[l].bk  = (const float*)d_in[s + 9];
        L[l].bm1 = (const float*)d_in[s + 10]; L[l].bm2 = (const float*)d_in[s + 11];
        L[l].bq  = (const float*)d_in[s + 12]; L[l].bv  = (const float*)d_in[s + 13];
    }
    const int Cdim[3] = {2, 8, 8};
    const int InD[3]  = {20, 32, 32};
    const int Cout[3] = {8, 8, 4};

    const long NN = 512L * 512L;
    float* w = (float*)d_ws;
    size_t off = 0;
    auto take = [&](size_t nfl) { float* p = w + off; off += nfl; return p; };
    float* xA     = take(8L * 512 * 32);
    float* xB     = take(8L * 512 * 32);
    float* adjcs0 = take(8L * 2 * NN);
    float* adjcs1 = take(8L * 8 * NN);
    float* adjcs2 = take(8L * 8 * NN);
    float* adjcs3 = take(8L * 4 * NN);
    float* adjn   = take(8L * 8 * NN);       // reused: adjn -> scores S -> adj-MLP tmp
    float* Abuf   = take(8L * 8 * NN);
    float* deg    = take(8L * 8 * 512);
    float* hq = take(8L * 8 * 512 * 32);
    float* hk = take(8L * 8 * 512 * 32);
    float* hv = take(8L * 8 * 512 * 32);
    float* Qb = take(8L * 8 * 512 * 32);
    float* Kb = take(8L * 8 * 512 * 32);
    float* Vb = take(8L * 8 * 512 * 32);

    float* adjcs[4] = { adjcs0, adjcs1, adjcs2, adjcs3 };

    k_build_x0<<<16, 256, 0, stream>>>(x, tt, xA);
    k_copy_ch0<<<8192, 256, 0, stream>>>(adj, adjcs0);
    // adjc channel 1 = adj @ adj  (bf16 is exact for 0/1 inputs, small counts)
    k_gemm_bf16<<<dim3(8, 32, 8), 128, 0, stream>>>(adj, adj, adjcs0 + NN,
        512, 512, 512, 512, NN, NN, 2 * NN);

    float* xin = xA; float* xout = xB;
    for (int l = 0; l < 3; ++l) {
        const int C = Cdim[l], in = InD[l], co = Cout[l];
        const int BC = 8 * C;
        const long tot = (long)BC * NN;
        k_degree<<<BC * 512 / 256, 256, 0, stream>>>(adjcs[l], deg, BC);
        k_normadj<<<(int)(tot / 256), 256, 0, stream>>>(adjcs[l], deg, adjn, tot);
        const int lint = BC * 512 * 32;
        k_linear<<<lint / 256, 256, 0, stream>>>(xin, L[l].Wq, L[l].bq, hq, C, in, lint);
        k_linear<<<lint / 256, 256, 0, stream>>>(xin, L[l].Wk, L[l].bk, hk, C, in, lint);
        k_linear<<<lint / 256, 256, 0, stream>>>(xin, L[l].Wv, L[l].bv, hv, C, in, lint);
        // GCN aggregation: (B*C) batched 512x512 @ 512x32, WMMA bf16
        k_gemm_bf16<<<dim3(8, 2, BC), 128, 0, stream>>>(adjn, hq, Qb, 512, 512, 32, 32, NN, 512L * 32, 512L * 32);
        k_gemm_bf16<<<dim3(8, 2, BC), 128, 0, stream>>>(adjn, hk, Kb, 512, 512, 32, 32, NN, 512L * 32, 512L * 32);
        k_gemm_bf16<<<dim3(8, 2, BC), 128, 0, stream>>>(adjn, hv, Vb, 512, 512, 32, 32, NN, 512L * 32, 512L * 32);
        k_scores<<<dim3(32, 32, BC), 32, 0, stream>>>(Qb, Kb, adjn);       // S into adjn
        k_sym<<<(int)(tot / 256), 256, 0, stream>>>(adjn, Abuf, 0.5f, tot);
        k_xmlp<<<16, 256, 0, stream>>>(Vb, L[l].Wm1, L[l].bm1, L[l].Wm2, L[l].bm2, xout, C);
        k_adjmlp<<<8192, 256, 0, stream>>>(Abuf, adjcs[l], L[l].W1, L[l].b1,
                                           L[l].W2, L[l].b2, adjn, C, co);  // tmp into adjn
        const long tot2 = 8L * co * NN;
        k_sym<<<(int)(tot2 / 256), 256, 0, stream>>>(adjn, adjcs[l + 1], 1.0f, tot2);
        float* sw = xin; xin = xout; xout = sw;
    }

    float* outp = (float*)d_out;
    k_final_adj<<<32768, 128, 0, stream>>>(adjcs0, adjcs1, adjcs2, adjcs3,
        faW1, fab1, faW2, fab2, faW3, fab3, outp + 8 * 512 * 10);
    k_final_x<<<16, 256, 0, stream>>>(xin, fxW1, fxb1, fxW2, fxb2, fxW3, fxb3, outp);
}
